// EnhancedGATGCN_7387343749409
// MI455X (gfx1250) — matrix operand order repaired and checked by reading, not compiled, not run on hardware
//
#include <hip/hip_runtime.h>
#include <hip/hip_bf16.h>

// ---------------------------------------------------------------------------
// Types for CDNA5 WMMA (wave32)
// ---------------------------------------------------------------------------
typedef __attribute__((ext_vector_type(16))) __bf16 v16bf;
typedef __attribute__((ext_vector_type(8)))  __bf16 v8bf;
typedef __attribute__((ext_vector_type(8)))  float  v8f;

#define BM 128
#define BN 64
#define BK 32

// Async global->LDS copy path (CDNA5), guarded so absence degrades cleanly.
// Builtin expects int4 pointers: global (AS1) source, LDS (AS3) destination.
#if defined(__AMDGCN__) && __has_builtin(__builtin_amdgcn_global_load_async_to_lds_b128) && __has_builtin(__builtin_amdgcn_s_wait_asynccnt)
#define USE_ASYNC 1
typedef int v4i_ __attribute__((ext_vector_type(4)));
typedef __attribute__((address_space(1))) v4i_ as1_v4i;
typedef __attribute__((address_space(3))) v4i_ as3_v4i;
#else
#define USE_ASYNC 0
#endif

// ---------------------------------------------------------------------------
// f32 -> bf16 converts (A gets zero-padded to Kp = roundup32(K))
// ---------------------------------------------------------------------------
__global__ void k_cvt_pad(const float* __restrict__ src, __bf16* __restrict__ dst,
                          int M, int K, int Kp) {
  size_t i = (size_t)blockIdx.x * blockDim.x + threadIdx.x;
  if (i >= (size_t)M * Kp) return;
  int m = (int)(i / Kp), k = (int)(i % Kp);
  dst[i] = (k < K) ? (__bf16)src[(size_t)m * K + k] : (__bf16)0.f;
}

__global__ void k_cvt(const float* __restrict__ src, __bf16* __restrict__ dst, size_t n) {
  size_t i = (size_t)blockIdx.x * blockDim.x + threadIdx.x;
  if (i < n) dst[i] = (__bf16)src[i];
}

// ---------------------------------------------------------------------------
// Tiled GEMM: C[M,N] = A[M,Kp(bf16)] @ B[K,N(bf16)] (+bias)(+relu), f32 out.
// Block = 256 threads = 8 waves; wave computes a 16(M) x 64(N) strip.
// A staged row-major [128][32]; B staged transposed [64][32] so each lane's
// WMMA fragment is one contiguous 32B LDS read.
// ---------------------------------------------------------------------------
__global__ __launch_bounds__(256) void k_gemm_bf16(
    const __bf16* __restrict__ A, int Kp,
    const __bf16* __restrict__ B,
    const float* __restrict__ bias, float* __restrict__ C,
    int M, int N, int K, int ldc, int relu)
{
  __shared__ __align__(32) __bf16 As[BM * BK];
  __shared__ __align__(32) __bf16 Bs[BN * BK];

  const int tid  = threadIdx.x;
  const int lane = tid & 31;
  const int wave = tid >> 5;
  const int bm = blockIdx.y * BM;
  const int bn = blockIdx.x * BN;

  v8f acc[4] = {};

  const int kb_a = (lane < 16) ? 0 : 16;   // A frag K base (contiguous k = 16h+s, matches B)
  const int kb_b = (lane < 16) ? 0 : 16;   // B frag K base (contiguous k = 16h+s, matches A)
  const int mloc = lane & 15;

  // staging coordinates (branch-free: clamped addresses + select-zero)
  const int ar  = tid >> 1;          // A row within tile, 0..127
  const int ah  = (tid & 1) * 16;    // A col offset {0,16}
  const int bnn = tid & 63;          // B col within tile, 0..63
  const int bkk = (tid >> 6) * 8;    // B k offset {0,8,16,24}
  const bool aok  = (bm + ar) < M;
  const int  arow = aok ? (bm + ar) : (M - 1);
  const bool nok  = (bn + bnn) < N;
  const int  bcol = nok ? (bn + bnn) : (N - 1);
#if USE_ASYNC
  const bool fullA = (bm + BM) <= M;   // uniform across block
#endif

  const int nk = Kp / BK;
  for (int kt = 0; kt < nk; ++kt) {
    const int k0 = kt * BK;

    // ---- A tile: 128x32 bf16, two 16B chunks per thread ----
#if USE_ASYNC
    if (fullA) {
      const __bf16* g = A + (size_t)(bm + ar) * Kp + k0 + ah;
      __bf16* l = &As[ar * BK + ah];
      __builtin_amdgcn_global_load_async_to_lds_b128((as1_v4i*)g,       (as3_v4i*)l,       0, 0);
      __builtin_amdgcn_global_load_async_to_lds_b128((as1_v4i*)(g + 8), (as3_v4i*)(l + 8), 0, 0);
    } else
#endif
    {
      const __bf16* g = A + (size_t)arow * Kp + k0 + ah;
      v8bf a0 = *(const v8bf*)g;
      v8bf a1 = *(const v8bf*)(g + 8);
      if (!aok) { v8bf z = {}; a0 = z; a1 = z; }
      *(v8bf*)&As[ar * BK + ah]     = a0;
      *(v8bf*)&As[ar * BK + ah + 8] = a1;
    }

    // ---- B tile: 32x64 bf16, transposed stage; 8 coalesced b16 loads ----
    {
      v8bf bt;
#pragma unroll
      for (int j = 0; j < 8; ++j) {
        int gk = k0 + bkk + j;
        int kc = (gk < K) ? gk : (K - 1);
        __bf16 v = B[(size_t)kc * N + bcol];
        bt[j] = ((gk < K) && nok) ? v : (__bf16)0.f;
      }
      *(v8bf*)&Bs[bnn * BK + bkk] = bt;   // one ds_store_b128
    }

#if USE_ASYNC
    if (fullA) __builtin_amdgcn_s_wait_asynccnt(0);
#endif
    __syncthreads();

    // ---- fragments ----
    union { v16bf v; v8bf h[2]; } af;
    const __bf16* ap = &As[(wave * 16 + mloc) * BK];
    af.h[0] = *(const v8bf*)(ap + kb_a);
    af.h[1] = *(const v8bf*)(ap + kb_a + 8);

    v16bf bf0 = *(const v16bf*)(&Bs[(0 * 16 + mloc) * BK + kb_b]);
    v16bf bf1 = *(const v16bf*)(&Bs[(1 * 16 + mloc) * BK + kb_b]);
    v16bf bf2 = *(const v16bf*)(&Bs[(2 * 16 + mloc) * BK + kb_b]);
    v16bf bf3 = *(const v16bf*)(&Bs[(3 * 16 + mloc) * BK + kb_b]);

    acc[0] = __builtin_amdgcn_wmma_f32_16x16x32_bf16(false, af.v, false, bf0, (short)0, acc[0], false, false);
    acc[1] = __builtin_amdgcn_wmma_f32_16x16x32_bf16(false, af.v, false, bf1, (short)0, acc[1], false, false);
    acc[2] = __builtin_amdgcn_wmma_f32_16x16x32_bf16(false, af.v, false, bf2, (short)0, acc[2], false, false);
    acc[3] = __builtin_amdgcn_wmma_f32_16x16x32_bf16(false, af.v, false, bf3, (short)0, acc[3], false, false);
    __syncthreads();
  }

  // ---- store: C/D layout: VGPR r -> M = r (lanes 0-15) / r+8 (lanes 16-31) ----
  const int row_base = bm + wave * 16 + ((lane < 16) ? 0 : 8);
#pragma unroll
  for (int t = 0; t < 4; ++t) {
    int gn = bn + t * 16 + mloc;
    if (gn >= N) continue;
    float bv = bias ? bias[gn] : 0.f;
#pragma unroll
    for (int r = 0; r < 8; ++r) {
      int gm = row_base + r;
      if (gm < M) {
        float v = acc[t][r] + bv;
        if (relu) v = fmaxf(v, 0.f);
        C[(size_t)gm * ldc + gn] = v;
      }
    }
  }
}

// ---------------------------------------------------------------------------
// Elementwise / scatter helpers
// ---------------------------------------------------------------------------
__global__ void k_fill(float* __restrict__ p, float v, size_t n) {
  size_t i = (size_t)blockIdx.x * blockDim.x + threadIdx.x;
  if (i < n) p[i] = v;
}

__device__ __forceinline__ void atomicMaxF(float* addr, float v) {
  if (v >= 0.f) atomicMax((int*)addr, __float_as_int(v));
  else          atomicMin((unsigned int*)addr, __float_as_uint(v));
}

__global__ void k_att_proj(const float* __restrict__ h, const float* __restrict__ a_src,
                           const float* __restrict__ a_dst, float* __restrict__ as_,
                           float* __restrict__ ad_, int n) {
  int i = blockIdx.x * blockDim.x + threadIdx.x;
  if (i >= n) return;
  int hd = i % 10;
  const float* hp = h + (size_t)(i / 10) * 780 + hd * 78;
  const float* s = a_src + hd * 78;
  const float* d = a_dst + hd * 78;
  float vs = 0.f, vd = 0.f;
  for (int f = 0; f < 78; ++f) { float x = hp[f]; vs += x * s[f]; vd += x * d[f]; }
  as_[i] = vs; ad_[i] = vd;
}

__global__ void k_edge_logits(const int* __restrict__ ei, const float* __restrict__ as_,
                              const float* __restrict__ ad_, float* __restrict__ ebuf,
                              float* __restrict__ m, int E, int N) {
  int i = blockIdx.x * blockDim.x + threadIdx.x;
  if (i >= (E + N) * 10) return;
  int e = i / 10, hd = i % 10;
  int s, d;
  if (e < E) { s = ei[e]; d = ei[E + e]; } else { s = d = e - E; }
  float v = as_[s * 10 + hd] + ad_[d * 10 + hd];
  v = (v > 0.f) ? v : 0.2f * v;
  ebuf[i] = v;
  atomicMaxF(&m[d * 10 + hd], v);
}

__global__ void k_edge_exp(const int* __restrict__ ei, const float* __restrict__ m,
                           float* __restrict__ ebuf, float* __restrict__ den, int E, int N) {
  int i = blockIdx.x * blockDim.x + threadIdx.x;
  if (i >= (E + N) * 10) return;
  int e = i / 10, hd = i % 10;
  int d = (e < E) ? ei[E + e] : (e - E);
  float ex = __expf(ebuf[i] - m[d * 10 + hd]);
  ebuf[i] = ex;
  atomicAdd(&den[d * 10 + hd], ex);
}

__global__ __launch_bounds__(256) void k_gat_agg(
    const int* __restrict__ ei, const float* __restrict__ h,
    const float* __restrict__ ebuf, const float* __restrict__ den,
    float* __restrict__ agg, int E) {
  int e = blockIdx.x;
  int s, d;
  if (e < E) { s = ei[e]; d = ei[E + e]; } else { s = d = e - E; }
  __shared__ float alpha[10];
  if (threadIdx.x < 10)
    alpha[threadIdx.x] = ebuf[e * 10 + threadIdx.x] / den[d * 10 + threadIdx.x];
  __syncthreads();
  const float* hs = h + (size_t)s * 780;
  float* ag = agg + (size_t)d * 780;
  for (int idx = threadIdx.x; idx < 780; idx += blockDim.x)
    atomicAdd(&ag[idx], hs[idx] * alpha[idx / 78]);
}

__global__ void k_bias_relu(float* __restrict__ x, const float* __restrict__ b,
                            size_t rows, int cols, int relu) {
  size_t i = (size_t)blockIdx.x * blockDim.x + threadIdx.x;
  if (i >= rows * cols) return;
  float v = x[i] + b[i % cols];
  x[i] = relu ? fmaxf(v, 0.f) : v;
}

__global__ void k_deg(const int* __restrict__ ei, float* __restrict__ deg, int E, int N) {
  int e = blockIdx.x * blockDim.x + threadIdx.x;
  if (e >= E + N) return;
  int d = (e < E) ? ei[E + e] : (e - E);
  atomicAdd(&deg[d], 1.f);
}

__global__ void k_dinv(float* __restrict__ deg, int N) {
  int i = blockIdx.x * blockDim.x + threadIdx.x;
  if (i < N) deg[i] = rsqrtf(fmaxf(deg[i], 1.f));
}

__global__ __launch_bounds__(256) void k_gcn_agg(
    const int* __restrict__ ei, const float* __restrict__ h2,
    const float* __restrict__ dinv, float* __restrict__ x2, int E) {
  int e = blockIdx.x;
  int s, d;
  if (e < E) { s = ei[e]; d = ei[E + e]; } else { s = d = e - E; }
  float norm = dinv[s] * dinv[d];
  const float* hp = h2 + (size_t)s * 780;
  float* xp = x2 + (size_t)d * 780;
  for (int idx = threadIdx.x; idx < 780; idx += blockDim.x)
    atomicAdd(&xp[idx], hp[idx] * norm);
}

__global__ void k_pool(const float* __restrict__ x2, const int* __restrict__ batch,
                       float* __restrict__ xg, int N) {
  size_t i = (size_t)blockIdx.x * blockDim.x + threadIdx.x;
  if (i >= (size_t)N * 780) return;
  int n = (int)(i / 780), j = (int)(i % 780);
  atomicAdd(&xg[(size_t)batch[n] * 780 + j], x2[i]);
}

// 1D conv over embedded sequence; emb table (26x128) resident in LDS.
__global__ __launch_bounds__(256) void k_conv(
    const int* __restrict__ target, const float* __restrict__ emb,
    const float* __restrict__ cW, const float* __restrict__ cb,
    float* __restrict__ c) {
  __shared__ float embs[26 * 128];
  __shared__ int tgt[16];
  int b = blockIdx.y;
  int o0 = blockIdx.x * 8;
  for (int i = threadIdx.x; i < 26 * 128; i += 256) embs[i] = emb[i];
  if (threadIdx.x < 16) {
    int oi = o0 + threadIdx.x;
    tgt[threadIdx.x] = (oi < 1000) ? target[b * 1000 + oi] : 0;
  }
  __syncthreads();
  int nf = threadIdx.x & 31;
  int oo = threadIdx.x >> 5;   // wave id -> lanes share one emb row (LDS bcast)
  int o = o0 + oo;
  if (o >= 993) return;
  float acc = cb[nf];
  const float* w = cW + nf * 1024;  // cW[nf][e][k] = cW[nf*1024 + e*8 + k]
#pragma unroll
  for (int k = 0; k < 8; ++k) {
    const float* er = &embs[tgt[oo + k] * 128];
#pragma unroll 16
    for (int e = 0; e < 128; ++e) acc += er[e] * w[e * 8 + k];
  }
  c[((size_t)b * 32 + nf) * 993 + o] = acc;
}

__global__ void k_out(const float* __restrict__ z4, const float* __restrict__ oW,
                      const float* __restrict__ ob, float* __restrict__ out) {
  int b = blockIdx.x * blockDim.x + threadIdx.x;
  if (b >= 200) return;
  float acc = ob[0];
  for (int j = 0; j < 128; ++j) acc += z4[b * 128 + j] * oW[j];
  out[b] = acc;
}

// ---------------------------------------------------------------------------
// Launch
// ---------------------------------------------------------------------------
extern "C" void kernel_launch(void* const* d_in, const int* in_sizes, int n_in,
                              void* d_out, int out_size, void* d_ws, size_t ws_size,
                              hipStream_t stream) {
  const int N = 20000, E = 400000, Bb = 200, HID = 780;
  const int Etot = E + N;

  const float* x        = (const float*)d_in[0];
  const int*   ei       = (const int*)  d_in[1];
  const int*   batch    = (const int*)  d_in[2];
  const int*   target   = (const int*)  d_in[3];
  const float* gat_W    = (const float*)d_in[4];
  const float* gat_asrc = (const float*)d_in[5];
  const float* gat_adst = (const float*)d_in[6];
  const float* gat_b    = (const float*)d_in[7];
  const float* gcn_W    = (const float*)d_in[8];
  const float* gcn_b    = (const float*)d_in[9];
  const float* fcg1_W   = (const float*)d_in[10];
  const float* fcg1_b   = (const float*)d_in[11];
  const float* fcg2_W   = (const float*)d_in[12];
  const float* fcg2_b   = (const float*)d_in[13];
  const float* emb      = (const float*)d_in[14];
  const float* cW       = (const float*)d_in[15];
  const float* cb       = (const float*)d_in[16];
  const float* fxt_W    = (const float*)d_in[17];
  const float* fxt_b    = (const float*)d_in[18];
  const float* f1_W = (const float*)d_in[19]; const float* f1_b = (const float*)d_in[20];
  const float* f2_W = (const float*)d_in[21]; const float* f2_b = (const float*)d_in[22];
  const float* f3_W = (const float*)d_in[23]; const float* f3_b = (const float*)d_in[24];
  const float* f4_W = (const float*)d_in[25]; const float* f4_b = (const float*)d_in[26];
  const float* o_W  = (const float*)d_in[27]; const float* o_b  = (const float*)d_in[28];
  float* out = (float*)d_out;

  float* W = (float*)d_ws;
  size_t off = 0;
  auto alloc = [&](size_t n) { size_t o = off; off += (n + 63) & ~size_t(63); return W + o; };
  const size_t BIG = (size_t)N * HID;        // 15.6M floats
  float* big1 = alloc(BIG);                  // h -> h2 -> conv output
  float* big2 = alloc(BIG);                  // agg/x1 -> x2
  float* a_s  = alloc((size_t)N * 10);
  float* a_d  = alloc((size_t)N * 10);
  float* mbuf = alloc((size_t)N * 10);
  float* den  = alloc((size_t)N * 10);
  float* ebuf = alloc((size_t)Etot * 10);
  float* deg  = alloc(N);
  float* xg   = alloc((size_t)Bb * HID);
  float* xg1  = alloc((size_t)Bb * 1500);
  float* xc   = alloc((size_t)Bb * 256);
  float* z1   = alloc((size_t)Bb * 1024);
  float* z2   = alloc((size_t)Bb * 512);
  float* z3   = alloc((size_t)Bb * 256);
  float* z4   = alloc((size_t)Bb * 128);
  __bf16* abf = (__bf16*)alloc(8 * 1000 * 1000);   // 16M bf16 (max 20000x800)
  __bf16* bbf = (__bf16*)alloc(2 * 1100 * 1000);   // 4.4M bf16 (max 31776x128)
  (void)ws_size; (void)in_sizes; (void)n_in; (void)out_size;

  auto cvtA = [&](const float* src, int M, int K) -> int {   // -> abf, returns Kp
    int Kp = (K + 31) & ~31;
    size_t n = (size_t)M * Kp;
    k_cvt_pad<<<(unsigned)((n + 255) / 256), 256, 0, stream>>>(src, abf, M, K, Kp);
    return Kp;
  };
  auto cvtB = [&](const float* src, size_t n) {              // -> bbf
    k_cvt<<<(unsigned)((n + 255) / 256), 256, 0, stream>>>(src, bbf, n);
  };
  auto gemm = [&](int Kp, const float* bias, float* Cm, int M, int Nn, int K,
                  int ldc, int relu) {
    dim3 g((Nn + BN - 1) / BN, (M + BM - 1) / BM);
    k_gemm_bf16<<<g, 256, 0, stream>>>(abf, Kp, bbf, bias, Cm, M, Nn, K, ldc, relu);
  };
  auto fill = [&](float* p, float v, size_t n) {
    k_fill<<<(unsigned)((n + 255) / 256), 256, 0, stream>>>(p, v, n);
  };

  // ---- GAT ----
  int Kp = cvtA(x, N, 78);
  cvtB(gat_W, (size_t)78 * HID);
  gemm(Kp, nullptr, big1, N, HID, 78, HID, 0);                              // h
  k_att_proj<<<(N * 10 + 255) / 256, 256, 0, stream>>>(big1, gat_asrc, gat_adst, a_s, a_d, N * 10);
  fill(mbuf, -3.0e38f, (size_t)N * 10);
  fill(den, 0.f, (size_t)N * 10);
  fill(deg, 0.f, N);
  k_edge_logits<<<(Etot * 10 + 255) / 256, 256, 0, stream>>>(ei, a_s, a_d, ebuf, mbuf, E, N);
  k_edge_exp<<<(Etot * 10 + 255) / 256, 256, 0, stream>>>(ei, mbuf, ebuf, den, E, N);
  fill(big2, 0.f, BIG);
  k_gat_agg<<<Etot, 256, 0, stream>>>(ei, big1, ebuf, den, big2, E);        // agg
  k_bias_relu<<<(unsigned)((BIG + 255) / 256), 256, 0, stream>>>(big2, gat_b, N, HID, 1); // x1

  // ---- GCN ----
  k_deg<<<(Etot + 255) / 256, 256, 0, stream>>>(ei, deg, E, N);
  k_dinv<<<(N + 255) / 256, 256, 0, stream>>>(deg, N);
  Kp = cvtA(big2, N, HID);
  cvtB(gcn_W, (size_t)HID * HID);
  gemm(Kp, nullptr, big1, N, HID, HID, HID, 0);                             // h2
  fill(big2, 0.f, BIG);
  k_gcn_agg<<<Etot, 256, 0, stream>>>(ei, big1, deg, big2, E);
  k_bias_relu<<<(unsigned)((BIG + 255) / 256), 256, 0, stream>>>(big2, gcn_b, N, HID, 1); // x2

  // ---- pool + graph MLP ----
  fill(xg, 0.f, (size_t)Bb * HID);
  k_pool<<<(unsigned)((BIG + 255) / 256), 256, 0, stream>>>(big2, batch, xg, N);
  Kp = cvtA(xg, Bb, HID);
  cvtB(fcg1_W, (size_t)HID * 1500);
  gemm(Kp, fcg1_b, xg1, Bb, 1500, HID, 1500, 1);
  Kp = cvtA(xg1, Bb, 1500);
  cvtB(fcg2_W, (size_t)1500 * 128);
  gemm(Kp, fcg2_b, xc, Bb, 128, 1500, 256, 0);                              // xc[:, :128]

  // ---- protein branch ----
  k_conv<<<dim3(125, Bb), 256, 0, stream>>>(target, emb, cW, cb, big1);     // c
  Kp = cvtA(big1, Bb, 32 * 993);
  cvtB(fxt_W, (size_t)32 * 993 * 128);
  gemm(Kp, fxt_b, xc + 128, Bb, 128, 32 * 993, 256, 0);                     // xc[:, 128:]

  // ---- fusion MLP ----
  Kp = cvtA(xc, Bb, 256); cvtB(f1_W, (size_t)256 * 1024);
  gemm(Kp, f1_b, z1, Bb, 1024, 256, 1024, 1);
  Kp = cvtA(z1, Bb, 1024); cvtB(f2_W, (size_t)1024 * 512);
  gemm(Kp, f2_b, z2, Bb, 512, 1024, 512, 1);
  Kp = cvtA(z2, Bb, 512); cvtB(f3_W, (size_t)512 * 256);
  gemm(Kp, f3_b, z3, Bb, 256, 512, 256, 1);
  Kp = cvtA(z3, Bb, 256); cvtB(f4_W, (size_t)256 * 128);
  gemm(Kp, f4_b, z4, Bb, 128, 256, 128, 1);
  k_out<<<1, 256, 0, stream>>>(z4, o_W, o_b, out);
}
